// HadamardExpansionV2_11192684773781
// MI455X (gfx1250) — matrix operand order repaired and checked by reading, not compiled, not run on hardware
//
#include <hip/hip_runtime.h>
#include <hip/hip_bf16.h>
#include <math.h>

// ---------------------------------------------------------------------------
// MI455X (gfx1250) implementation.
//   B=16, C1=512, H=W=32 (HW=1024), CS=32, PAIRS=496, OUTC=1008.
// GEMM (1x1 conv) runs on v_wmma_f32_16x16x32_bf16 with split-bf16 (hi/lo)
// operands for ~fp32 accuracy: w*x ~= wh*xh + wh*xl + wl*xh (fp32 accum).
// X k-slabs are staged to LDS by the Tensor Data Mover (tensor_load_to_lds)
// in an explicit double-buffered pipeline synchronized with s_wait_tensorcnt.
// ---------------------------------------------------------------------------

#define C1 512
#define HW 1024
#define NB 16
#define CS 32
#define PAIRS 496
#define OUTC 1008
#define EPSV 1e-5f

typedef __attribute__((ext_vector_type(16))) __bf16 v16bf;
typedef __attribute__((ext_vector_type(8)))  float  v8f;
typedef __attribute__((ext_vector_type(4)))  unsigned int u32x4;
typedef __attribute__((ext_vector_type(8)))  int          i32x8;
typedef __attribute__((ext_vector_type(4)))  int          i32x4;

#if defined(__HIP_DEVICE_COMPILE__) && \
    __has_builtin(__builtin_amdgcn_tensor_load_to_lds) && \
    __has_builtin(__builtin_amdgcn_s_wait_tensorcnt)
#define USE_TDM 1
#else
#define USE_TDM 0
#endif

// Decisive probe: fail the device compile loudly if the TDM builtins are not
// actually available, so "compiled OK" from here on PROVES tensor_load_to_lds
// is in the emitted gfx1250 code (no silent fallback).
#if defined(__HIP_DEVICE_COMPILE__) && defined(__gfx1250__) && !USE_TDM
#error "gfx1250 device pass lacks tensor_load_to_lds / s_wait_tensorcnt builtins"
#endif

__device__ __forceinline__ void pair_decode(int q, int& i, int& j) {
    // np.triu_indices(32, k=1) order: i ascending, j = i+1..31
    int i0 = 0, rem = q;
    while (rem >= 31 - i0) { rem -= 31 - i0; ++i0; }
    i = i0;
    j = i0 + 1 + rem;
}

#if USE_TDM
// Issue a TDM copy of a 2D fp32 tile [32 rows x 128 cols] whose row stride in
// memory is HW floats, into LDS at byte address lds_addr.
// D# layout per CDNA5 ISA sec 8.3/8.4 (group0: count/lds/global/type,
// group1: data_size/tensor dims/tile dims/stride). Groups 2/3 zero (2D).
__device__ __forceinline__ void tdm_load_slab(const float* gsrc, unsigned lds_addr)
{
    unsigned long long ga = (unsigned long long)(size_t)gsrc;
    u32x4 g0;
    g0[0] = 1u;                                        // count=1, user desc
    g0[1] = lds_addr;                                  // LDS byte address
    g0[2] = (unsigned)(ga & 0xFFFFFFFFu);              // global_addr[31:0]
    g0[3] = (unsigned)((ga >> 32) & 0x01FFFFFFu)       // global_addr[56:32]
          | (2u << 30);                                // type = 2 ("image")
    i32x8 g1;
    g1[0] = (int)(2u << 16);          // data_size=2 (4 bytes); mask/pad/iter=0
    g1[1] = (int)(1024u << 16);       // tensor_dim0[15:0]=1024 (bits 63:48)
    g1[2] = (int)(512u  << 16);       // tensor_dim0 hi=0 | tensor_dim1[15:0]=512
    g1[3] = (int)(128u  << 16);       // tensor_dim1 hi=0 | tile_dim0=128
    g1[4] = 32;                       // tile_dim1=32 | tile_dim2=0
    g1[5] = 1024;                     // tensor_dim0_stride[31:0]=1024
    g1[6] = 0;                        // stride hi | tensor_dim1_stride lo
    g1[7] = 0;                        // tensor_dim1_stride hi
    i32x4 gz = {0, 0, 0, 0};
#if __clang_major__ >= 23
    i32x8 gz8 = {0, 0, 0, 0, 0, 0, 0, 0};
    __builtin_amdgcn_tensor_load_to_lds(g0, g1, gz, gz, gz8, 0);
#else
    __builtin_amdgcn_tensor_load_to_lds(g0, g1, gz, gz, 0);
#endif
}
#endif

// One K=32 MMA step: load/convert A from global, gather/convert B from LDS,
// then 6 split-precision WMMAs (A reused across both N-subtiles).
__device__ __forceinline__ void gemm_step(const float* __restrict__ xs,
                                          const float* __restrict__ w,
                                          int o0, int k0, int mrow, int kg, int n0,
                                          v8f& acc0, v8f& acc1)
{
    // ---- A tile (fc_w rows), split bf16. ISA 16-bit 16x32 A layout:
    // lanes 0-15: K = {kg8..kg8+7, 16+kg8..23+kg8}, kg8 = (lane>>4)*8
    const float4* wr4 = (const float4*)(w + (size_t)(o0 + mrow) * C1 + k0 + kg * 8);
    float4 wa = wr4[0], wb = wr4[1], wc = wr4[4], wd = wr4[5];
    float wf[16] = {wa.x, wa.y, wa.z, wa.w, wb.x, wb.y, wb.z, wb.w,
                    wc.x, wc.y, wc.z, wc.w, wd.x, wd.y, wd.z, wd.w};
    v16bf a_hi, a_lo;
    #pragma unroll
    for (int jj = 0; jj < 16; ++jj) {
        float v  = wf[jj];
        __bf16 h = (__bf16)v;
        a_hi[jj] = h;
        a_lo[jj] = (__bf16)(v - (float)h);
    }

    // ---- B tiles from LDS. 16-bit 32x16 B layout (same K order as A):
    // column n = lane&15, element j -> K = (lane>>4)*8 + (j&7) + (j>>3)*16
    v16bf b0_hi, b0_lo, b1_hi, b1_lo;
    #pragma unroll
    for (int jj = 0; jj < 16; ++jj) {
        const float* row = xs + (kg * 8 + (jj & 7) + (jj >> 3) * 16) * 128;
        float v0 = row[n0];
        float v1 = row[n0 + 16];
        __bf16 h0 = (__bf16)v0, h1 = (__bf16)v1;
        b0_hi[jj] = h0;
        b0_lo[jj] = (__bf16)(v0 - (float)h0);
        b1_hi[jj] = h1;
        b1_lo[jj] = (__bf16)(v1 - (float)h1);
    }

    // ---- split-precision MMA: hi*hi + lo*hi + hi*lo (fp32 accumulate) ----
    acc0 = __builtin_amdgcn_wmma_f32_16x16x32_bf16(false, a_hi, false, b0_hi,
                                                   (short)0, acc0, false, false);
    acc0 = __builtin_amdgcn_wmma_f32_16x16x32_bf16(false, a_lo, false, b0_hi,
                                                   (short)0, acc0, false, false);
    acc0 = __builtin_amdgcn_wmma_f32_16x16x32_bf16(false, a_hi, false, b0_lo,
                                                   (short)0, acc0, false, false);
    acc1 = __builtin_amdgcn_wmma_f32_16x16x32_bf16(false, a_hi, false, b1_hi,
                                                   (short)0, acc1, false, false);
    acc1 = __builtin_amdgcn_wmma_f32_16x16x32_bf16(false, a_lo, false, b1_hi,
                                                   (short)0, acc1, false, false);
    acc1 = __builtin_amdgcn_wmma_f32_16x16x32_bf16(false, a_hi, false, b1_lo,
                                                   (short)0, acc1, false, false);
}

// ---------------------------------------------------------------------------
// Kernel 1: fused 1x1-conv GEMM + bias + BatchNorm epilogue.
// Grid: (HW/128, C1/16, B); block = 128 threads = 4 wave32 waves.
// Block computes out[b, o0:o0+16, p0:p0+128]; wave w owns 32 columns via
// two accumulators. TDM double-buffers the X slab (12 WMMAs per 2 K-steps).
// ---------------------------------------------------------------------------
__global__ __launch_bounds__(128)
void conv_bn_gemm(const float* __restrict__ x,  const float* __restrict__ w,
                  const float* __restrict__ fcb,
                  const float* __restrict__ gamma, const float* __restrict__ beta,
                  const float* __restrict__ mean,  const float* __restrict__ var,
                  float* __restrict__ out)
{
    __shared__ float Xs[2][32 * 128];          // double-buffered K-slab of X

    const int b    = blockIdx.z;
    const int o0   = blockIdx.y * 16;
    const int p0   = blockIdx.x * 128;
    const int t    = threadIdx.x;
    const int wv   = t >> 5;                   // wave 0..3 (wave32)
    const int lane = t & 31;
    const int mrow = lane & 15;                // A-layout row for this lane
    const int kg   = lane >> 4;                // lane half selects K group
    const int n0   = wv * 32 + (lane & 15);

    const float* xb = x + (size_t)b * C1 * HW;

    v8f acc0 = {}, acc1 = {};

#if USE_TDM
    const unsigned lds0 = (unsigned)(size_t)&Xs[0][0];
    const unsigned lds1 = (unsigned)(size_t)&Xs[1][0];
    if (wv == 0) tdm_load_slab(xb + p0, lds0);

    for (int k0 = 0; k0 < C1; k0 += 64) {
        // ---- half-step A: consume buf0, async-fill buf1 ----
        __builtin_amdgcn_s_wait_tensorcnt(0);  // buf0 in LDS (no-op for wv>0)
        __syncthreads();                       // publish buf0 / retire buf1 readers
        if (wv == 0)
            tdm_load_slab(xb + (size_t)(k0 + 32) * HW + p0, lds1);
        gemm_step(&Xs[0][0], w, o0, k0, mrow, kg, n0, acc0, acc1);

        // ---- half-step B: consume buf1, async-fill buf0 ----
        __builtin_amdgcn_s_wait_tensorcnt(0);  // buf1 in LDS
        __syncthreads();                       // publish buf1 / retire buf0 readers
        if (wv == 0 && k0 + 64 < C1)
            tdm_load_slab(xb + (size_t)(k0 + 64) * HW + p0, lds0);
        gemm_step(&Xs[1][0], w, o0, k0 + 32, mrow, kg, n0, acc0, acc1);
    }
#else
    for (int k0 = 0; k0 < C1; k0 += 32) {
        // fallback staging: batched float4 loads -> ds_store_b128
        float4 rg[8];
        #pragma unroll
        for (int e = 0; e < 8; ++e) {
            int fi = t * 32 + e * 4;
            int kk = fi >> 7, pp = fi & 127;
            rg[e] = *(const float4*)&xb[(size_t)(k0 + kk) * HW + p0 + pp];
        }
        __syncthreads();
        #pragma unroll
        for (int e = 0; e < 8; ++e)
            *(float4*)&Xs[0][t * 32 + e * 4] = rg[e];
        __syncthreads();
        gemm_step(&Xs[0][0], w, o0, k0, mrow, kg, n0, acc0, acc1);
    }
#endif

    // ---- epilogue: + fc_b, BatchNorm(eval), store into out[:, 0:512] ----
    // C/D layout: VGPR v, lanes 0-15 -> M=v, N=lane; lanes 16-31 -> M=v+8.
    const int col = lane & 15;
    #pragma unroll
    for (int v = 0; v < 8; ++v) {
        int o = o0 + v + 8 * kg;
        float s  = gamma[o] * rsqrtf(var[o] + EPSV);
        float bb = beta[o] - (mean[o] - fcb[o]) * s;
        size_t base = ((size_t)b * OUTC + o) * HW + p0 + wv * 32 + col;
        out[base]      = acc0[v] * s + bb;
        out[base + 16] = acc1[v] * s + bb;
    }
}

// ---------------------------------------------------------------------------
// Kernel 2: global average pool over HW. Grid (C1, B), block 256.
// ---------------------------------------------------------------------------
__global__ __launch_bounds__(256)
void pool_kernel(const float* __restrict__ out, float* __restrict__ pooled)
{
    const int c = blockIdx.x, b = blockIdx.y;
    const float* p = out + ((size_t)b * OUTC + c) * HW;
    float s = 0.f;
    for (int i = threadIdx.x; i < HW; i += 256) s += p[i];
    __shared__ float red[256];
    red[threadIdx.x] = s;
    __syncthreads();
    for (int off = 128; off > 0; off >>= 1) {
        if (threadIdx.x < off) red[threadIdx.x] += red[threadIdx.x + off];
        __syncthreads();
    }
    if (threadIdx.x == 0) pooled[b * C1 + c] = red[0] * (1.0f / (float)HW);
}

// ---------------------------------------------------------------------------
// Kernel 3: logits = pooled @ eva_w.T + eva_b. Grid (2, B), block 256.
// ---------------------------------------------------------------------------
__global__ __launch_bounds__(256)
void logits_kernel(const float* __restrict__ pooled, const float* __restrict__ evaw,
                   const float* __restrict__ evab, float* __restrict__ logits)
{
    const int o = blockIdx.x * 256 + threadIdx.x;
    const int b = blockIdx.y;
    const float* pv = pooled + b * C1;
    const float* wr = evaw + (size_t)o * C1;
    float s = evab[o];
    for (int c = 0; c < C1; ++c) s += pv[c] * wr[c];
    logits[b * C1 + o] = s;
}

// ---------------------------------------------------------------------------
// Kernel 4: top-32 per batch (descending; strict '>' matches JAX tie-break).
// Tiny work (16 x 512 x 32) -> one thread per batch, deterministic.
// ---------------------------------------------------------------------------
__global__ void topk_kernel(const float* __restrict__ logits, int* __restrict__ idx)
{
    const int b = blockIdx.x;
    const float* lg = logits + b * C1;
    float v[C1];
    for (int c = 0; c < C1; ++c) v[c] = lg[c];
    for (int s = 0; s < CS; ++s) {
        float best = -INFINITY;
        int bi = 0;
        for (int c = 0; c < C1; ++c)
            if (v[c] > best) { best = v[c]; bi = c; }
        idx[b * CS + s] = bi;
        v[bi] = -INFINITY;
    }
}

// ---------------------------------------------------------------------------
// Kernel 5: per-(pair,batch) partial sums of prod and prod^2 (deterministic).
// Grid (PAIRS, B), block 256.
// ---------------------------------------------------------------------------
__global__ __launch_bounds__(256)
void pair_stats(const float* __restrict__ out, const int* __restrict__ idx,
                float* __restrict__ psum, float* __restrict__ psq)
{
    const int q = blockIdx.x, b = blockIdx.y;
    int i, j; pair_decode(q, i, j);
    const int ci = idx[b * CS + i], cj = idx[b * CS + j];
    const float* xi = out + ((size_t)b * OUTC + ci) * HW;
    const float* xj = out + ((size_t)b * OUTC + cj) * HW;
    float s = 0.f, ss = 0.f;
    for (int t = threadIdx.x; t < HW; t += 256) {
        float p = xi[t] * xj[t];
        s += p; ss += p * p;
    }
    __shared__ float rs[256], rq[256];
    rs[threadIdx.x] = s; rq[threadIdx.x] = ss;
    __syncthreads();
    for (int off = 128; off > 0; off >>= 1) {
        if (threadIdx.x < off) {
            rs[threadIdx.x] += rs[threadIdx.x + off];
            rq[threadIdx.x] += rq[threadIdx.x + off];
        }
        __syncthreads();
    }
    if (threadIdx.x == 0) { psum[q * NB + b] = rs[0]; psq[q * NB + b] = rq[0]; }
}

// ---------------------------------------------------------------------------
// Kernel 6: finalize per-pair scale/offset: out = prod*sc + off.
// sc = g / sqrt(var+eps), off = b - mean*sc, g/b from bn params at (hi,hj).
// ---------------------------------------------------------------------------
__global__ __launch_bounds__(256)
void pair_finalize(const float* __restrict__ psum, const float* __restrict__ psq,
                   const float* __restrict__ gamma, const float* __restrict__ beta,
                   float* __restrict__ sc, float* __restrict__ off)
{
    const int q = blockIdx.x * 256 + threadIdx.x;
    if (q >= PAIRS) return;
    float s = 0.f, ss = 0.f;
    for (int b = 0; b < NB; ++b) { s += psum[q * NB + b]; ss += psq[q * NB + b]; }
    const float inv_n = 1.0f / (float)(NB * HW);
    float mean = s * inv_n;
    float varv = ss * inv_n - mean * mean;
    float rsig = rsqrtf(varv + EPSV);
    int i, j; pair_decode(q, i, j);
    float g  = gamma[i] * gamma[j];
    float bb = beta[i] * beta[j];
    sc[q]  = rsig * g;
    off[q] = bb - mean * rsig * g;
}

// ---------------------------------------------------------------------------
// Kernel 7: recompute products, normalize, write out[:, 512:1008].
// Grid (PAIRS, B), block 256.
// ---------------------------------------------------------------------------
__global__ __launch_bounds__(256)
void pair_write(const float* __restrict__ outr, const int* __restrict__ idx,
                const float* __restrict__ sc, const float* __restrict__ off,
                float* __restrict__ out)
{
    const int q = blockIdx.x, b = blockIdx.y;
    int i, j; pair_decode(q, i, j);
    const int ci = idx[b * CS + i], cj = idx[b * CS + j];
    const float* xi = outr + ((size_t)b * OUTC + ci) * HW;
    const float* xj = outr + ((size_t)b * OUTC + cj) * HW;
    const float s = sc[q], o = off[q];
    float* dst = out + ((size_t)b * OUTC + C1 + q) * HW;
    for (int t = threadIdx.x; t < HW; t += 256)
        dst[t] = xi[t] * xj[t] * s + o;
}

// ---------------------------------------------------------------------------
extern "C" void kernel_launch(void* const* d_in, const int* in_sizes, int n_in,
                              void* d_out, int out_size, void* d_ws, size_t ws_size,
                              hipStream_t stream)
{
    const float* x     = (const float*)d_in[0];
    const float* fcw   = (const float*)d_in[1];
    const float* fcb   = (const float*)d_in[2];
    const float* gamma = (const float*)d_in[3];
    const float* beta  = (const float*)d_in[4];
    const float* mean  = (const float*)d_in[5];
    const float* var   = (const float*)d_in[6];
    const float* evaw  = (const float*)d_in[7];
    const float* evab  = (const float*)d_in[8];

    float* out = (float*)d_out;
    float* ws  = (float*)d_ws;

    // workspace layout (float elements)
    float* pooled = ws;                        // 16*512      = 8192
    float* logits = ws + 8192;                 // 16*512      = 8192
    int*   idxp   = (int*)(ws + 16384);        // 16*32 ints  = 512
    float* psum   = ws + 16896;                // 496*16      = 7936
    float* psq    = ws + 24832;                // 496*16      = 7936
    float* scv    = ws + 32768;                // 496
    float* offv   = ws + 33264;                // 496  (total ~135 KB)

    dim3 g1(HW / 128, C1 / 16, NB);
    conv_bn_gemm<<<g1, 128, 0, stream>>>(x, fcw, fcb, gamma, beta, mean, var, out);

    dim3 g2(C1, NB);
    pool_kernel<<<g2, 256, 0, stream>>>(out, pooled);

    dim3 g3(C1 / 256, NB);
    logits_kernel<<<g3, 256, 0, stream>>>(pooled, evaw, evab, logits);

    topk_kernel<<<NB, 1, 0, stream>>>(logits, idxp);

    dim3 g5(PAIRS, NB);
    pair_stats<<<g5, 256, 0, stream>>>(out, idxp, psum, psq);

    pair_finalize<<<(PAIRS + 255) / 256, 256, 0, stream>>>(psum, psq, gamma, beta, scv, offv);

    pair_write<<<g5, 256, 0, stream>>>(out, idxp, scv, offv, out);
}
